// AxialAttention_wopos_5669356834379
// MI455X (gfx1250) — matrix operand order repaired and checked by reading, not compiled, not run on hardware
//
#include <hip/hip_runtime.h>
#include <hip/hip_bf16.h>

// AxialAttention (no positional) pipeline for MI455X (gfx1250, wave32).
// Memory-bound (~1.3 GB pipeline traffic @ 23.3 TB/s); all matmuls use
// v_wmma_f32_16x16x32_f16 (f16 multiplicands, f32 accumulate). BN applies
// are fused into consumers; BN statistics are cross-workgroup reductions
// via an atomics-based stats kernel. qkv-weight is pre-converted to f16 so
// WMMA A operands load as two b128s (the ISA 16-bit A layout is two
// contiguous 8-half runs per lane). k_qk stages its q/k slab with
// GLOBAL_LOAD_ASYNC_TO_LDS_B128 (ASYNCcnt) when the toolchain exposes it.

typedef __attribute__((ext_vector_type(16))) _Float16 v16h;
typedef __attribute__((ext_vector_type(8)))  _Float16 v8h;
typedef __attribute__((ext_vector_type(8)))  float    v8f;

#define BN_EPS 1e-5f

#if __has_builtin(__builtin_amdgcn_global_load_async_to_lds_b128) && \
    __has_builtin(__builtin_amdgcn_s_wait_asynccnt)
#define HAS_ASYNC_LDS 1
// Pointer types per the builtin's signature: v4i in addrspace(1)/(3).
typedef __attribute__((__vector_size__(4 * sizeof(int)))) int v4i_t;
typedef __attribute__((address_space(1))) v4i_t* gv4i_p;
typedef __attribute__((address_space(3))) v4i_t* lv4i_p;
#else
#define HAS_ASYNC_LDS 0
#endif

__device__ __forceinline__ v8f wmma16(v16h a, v16h b, v8f c) {
  // (neg_a, A, neg_b, B, c_mod, C, reuse_a, reuse_b)
  return __builtin_amdgcn_wmma_f32_16x16x32_f16(false, a, false, b, (short)0, c,
                                                false, false);
}

// B operand: per-lane n = lane&15, kgrp = lane>>4; 16 contiguous halves from
// a [n][k] row-major f16 tile -> two 16-byte loads.
__device__ __forceinline__ v16h load_b_rowk(const _Float16* p) {
  v8h lo = *(const v8h*)p;
  v8h hi = *(const v8h*)(p + 8);
  return __builtin_shufflevector(lo, hi, 0, 1, 2, 3, 4, 5, 6, 7,
                                 8, 9, 10, 11, 12, 13, 14, 15);
}

// A operand: per lane the K indices are one contiguous 16-half run
// [16*half, 16*half+16) (matching the B operand order). Row points at tile[m][koff]
// of a row-major f16 matrix (LDS or global); koff must be 16B-aligned.
__device__ __forceinline__ v16h load_a_f16(const _Float16* row, int half) {
  v8h lo = *(const v8h*)(row + 16 * half);
  v8h hi = *(const v8h*)(row + 16 * half + 8);
  return __builtin_shufflevector(lo, hi, 0, 1, 2, 3, 4, 5, 6, 7,
                                 8, 9, 10, 11, 12, 13, 14, 15);
}

__device__ __forceinline__ void bn_coef(const float* __restrict__ stats,
                                        const float* __restrict__ gam,
                                        const float* __restrict__ bet, int ch,
                                        float cnt, float* sc, float* sh) {
  const float m   = stats[2 * ch] / cnt;
  const float var = stats[2 * ch + 1] / cnt - m * m;
  const float s   = gam[ch] * rsqrtf(var + BN_EPS);
  *sc = s;
  *sh = bet[ch] - m * s;
}

// ---------------------------------------------------------------------------
// Kernel 0: zero the BN stats accumulators (sum, sumsq interleaved).
__global__ void k_zero(float* __restrict__ p, int n) {
  const int i = blockIdx.x * blockDim.x + threadIdx.x;
  if (i < n) p[i] = 0.f;
}

// ---------------------------------------------------------------------------
// Kernel 0b: pre-convert w_qkv (1024x512 f32) to f16 row-major.
__global__ __launch_bounds__(256) void k_cvt_w(const float* __restrict__ wq,
                                               _Float16* __restrict__ wh,
                                               int n) {
  const int i = blockIdx.x * 256 + threadIdx.x;
  if (i < n) wh[i] = (_Float16)wq[i];
}

// ---------------------------------------------------------------------------
// Kernel 1: qkv[b][o][h] = sum_c w_qkv[o][c] * x[n][c][h][w],  b = n*64 + w.
// One block per b; X_b (512x64) staged as f16 in LDS as [h][c] (the WMMA B
// layout); each of 8 waves owns 8 o-tiles; K=512 -> 16 WMMA steps per tile.
// A operands come from the pre-converted f16 weight (two b128s per step).
__global__ __launch_bounds__(256) void k_qkv_gemm(
    const float* __restrict__ x, const _Float16* __restrict__ wh,
    float* __restrict__ qkv_raw) {
  __shared__ __align__(16) _Float16 xs[64][512];  // [h][c], 64 KB
  const int b    = blockIdx.x;  // 0..1023
  const int nimg = b >> 6, w = b & 63;
  const int tid  = threadIdx.x;

  for (int e = tid; e < 64 * 512; e += 256) {
    const int c = e >> 6, h = e & 63;
    xs[h][c] = (_Float16)x[(((size_t)nimg * 512 + c) * 64 + h) * 64 + w];
  }
  __syncthreads();

  const int wave = tid >> 5, lane = tid & 31;
  const int half = lane >> 4, ln = lane & 15, mh = half * 8;
  float* outb = qkv_raw + (size_t)b * 65536;

  for (int ot = wave; ot < 64; ot += 8) {
    const int o0 = ot * 16;
    const _Float16* arow = wh + (size_t)(o0 + ln) * 512;
    v8f acc0 = {}, acc1 = {}, acc2 = {}, acc3 = {};
    for (int kk = 0; kk < 16; ++kk) {
      const int koff = kk * 32;
      const v16h a  = load_a_f16(arow + koff, half);
      const v16h b0 = load_b_rowk(&xs[0 + ln][koff + half * 16]);
      const v16h b1 = load_b_rowk(&xs[16 + ln][koff + half * 16]);
      const v16h b2 = load_b_rowk(&xs[32 + ln][koff + half * 16]);
      const v16h b3 = load_b_rowk(&xs[48 + ln][koff + half * 16]);
      acc0 = wmma16(a, b0, acc0);
      acc1 = wmma16(a, b1, acc1);
      acc2 = wmma16(a, b2, acc2);
      acc3 = wmma16(a, b3, acc3);
    }
#pragma unroll
    for (int r = 0; r < 8; ++r) {
      const size_t mo = (size_t)(o0 + mh + r) * 64;
      outb[mo + 0 + ln]  = acc0[r];
      outb[mo + 16 + ln] = acc1[r];
      outb[mo + 32 + ln] = acc2[r];
      outb[mo + 48 + ln] = acc3[r];
    }
  }
}

// ---------------------------------------------------------------------------
// Kernel 2 (generic): per-channel sum / sumsq over (outer x inner) elements.
// addr = o*outer_stride + ch*ch_stride + i ; grid = (channels, nsplit).
__global__ __launch_bounds__(256) void k_bn_stats(
    const float* __restrict__ src, float* __restrict__ stats, int inner_shift,
    int ch_stride, int outer_stride, int total_per_ch) {
  const int ch = blockIdx.x;
  const int per = total_per_ch / gridDim.y;
  const int start = blockIdx.y * per;
  const int imask = (1 << inner_shift) - 1;
  const int tid = threadIdx.x;
  float s1 = 0.f, s2 = 0.f;
  for (int e = start + tid; e < start + per; e += 256) {
    const int o = e >> inner_shift, i = e & imask;
    const float v =
        src[(size_t)o * outer_stride + (size_t)ch * ch_stride + i];
    s1 += v;
    s2 += v * v;
  }
  __shared__ float r1[256], r2[256];
  r1[tid] = s1;
  r2[tid] = s2;
  __syncthreads();
  for (int s = 128; s > 0; s >>= 1) {
    if (tid < s) {
      r1[tid] += r1[tid + s];
      r2[tid] += r2[tid + s];
    }
    __syncthreads();
  }
  if (tid == 0) {
    atomicAdd(&stats[2 * ch], r1[0]);
    atomicAdd(&stats[2 * ch + 1], r2[0]);
  }
}

// ---------------------------------------------------------------------------
// Kernel 3: qk[bg][i][j] = sum_c q[c][i]*k[c][j] with fused qkv-BN.
// q = channels g*128+[0,32), k = g*128+[32,64). K=32 -> one WMMA per tile.
// The contiguous 16 KB q/k slab is staged into LDS with async B128 loads
// (ASYNCcnt path) when available, then BN-applied/transposed to f16 tiles.
__global__ __launch_bounds__(256) void k_qk(
    const float* __restrict__ qkv_raw, const float* __restrict__ st_qkv,
    const float* __restrict__ g_qkv, const float* __restrict__ b_qkv,
    float* __restrict__ qk_raw) {
  __shared__ __align__(16) _Float16 qT[64][32];  // [i][c]  (A layout)
  __shared__ __align__(16) _Float16 kT[64][32];  // [j][c]  (B layout)
  __shared__ __align__(16) float stg[64][64];    // raw q|k rows (cc, h)
  const int bg = blockIdx.x;
  const int bidx = bg >> 3, g = bg & 7;
  const int tid = threadIdx.x;
  const float* src =
      qkv_raw + (size_t)bidx * 65536 + (size_t)g * 128 * 64;  // cc=0..63 slab

#if HAS_ASYNC_LDS
  // 4096 contiguous floats -> 1024 per-lane b128 transfers (4 per thread).
  float* stg0 = &stg[0][0];
  for (int t4 = tid; t4 < 1024; t4 += 256) {
    __builtin_amdgcn_global_load_async_to_lds_b128(
        (gv4i_p)(src + (size_t)t4 * 4), (lv4i_p)(stg0 + t4 * 4), 0, 0);
  }
  __builtin_amdgcn_s_wait_asynccnt(0);
  __syncthreads();
#else
  for (int e = tid; e < 4096; e += 256) stg[e >> 6][e & 63] = src[e];
  __syncthreads();
#endif

  // BN-apply + f16 convert into the transposed WMMA tiles.
  for (int e = tid; e < 4096; e += 256) {
    const int cc = e >> 6, h = e & 63;
    const int o = g * 128 + cc;
    float sc, sh;
    bn_coef(st_qkv, g_qkv, b_qkv, o, 65536.f, &sc, &sh);
    const float v = stg[cc][h] * sc + sh;
    if (cc < 32)
      qT[h][cc] = (_Float16)v;
    else
      kT[h][cc - 32] = (_Float16)v;
  }
  __syncthreads();

  const int wave = tid >> 5, lane = tid & 31;
  const int half = lane >> 4, ln = lane & 15, mh = half * 8;
  float* outbg = qk_raw + (size_t)bg * 4096;
  for (int t = wave; t < 16; t += 8) {
    const int i0 = (t >> 2) * 16, j0 = (t & 3) * 16;
    const v16h a  = load_a_f16(&qT[i0 + ln][0], half);
    const v16h bm = load_b_rowk(&kT[j0 + ln][half * 16]);
    v8f acc = {};
    acc = wmma16(a, bm, acc);
#pragma unroll
    for (int r = 0; r < 8; ++r)
      outbg[(size_t)(i0 + mh + r) * 64 + j0 + ln] = acc[r];
  }
}

// ---------------------------------------------------------------------------
// Kernel 4: sim = softmax(BN_g(qk)); sv[c][i] = sum_j v[c][j]*sim[i][j].
// v = channels g*128+[64,128) with fused qkv-BN. K=64 -> 2 WMMA per tile.
__global__ __launch_bounds__(256) void k_softmax_sv(
    const float* __restrict__ qkv_raw, const float* __restrict__ qk_raw,
    const float* __restrict__ st_qkv, const float* __restrict__ g_qkv,
    const float* __restrict__ b_qkv, const float* __restrict__ st_qk,
    const float* __restrict__ g_sim, const float* __restrict__ b_sim,
    float* __restrict__ sv_raw) {
  __shared__ __align__(16) _Float16 vrows[64][64];  // [c][j] (A layout)
  __shared__ __align__(16) _Float16 simh[64][64];   // [i][j] (B layout)
  __shared__ float srow[64][65];                    // padded f32 scratch
  const int bg = blockIdx.x;
  const int bidx = bg >> 3, g = bg & 7;
  const int tid = threadIdx.x;

  float gsc, gsh;
  bn_coef(st_qk, g_sim, b_sim, g, 1024.f * 4096.f, &gsc, &gsh);

  for (int e = tid; e < 4096; e += 256) {
    const int c = e >> 6, j = e & 63;
    const int o = g * 128 + 64 + c;
    float sc, sh;
    bn_coef(st_qkv, g_qkv, b_qkv, o, 65536.f, &sc, &sh);
    vrows[c][j] = (_Float16)(
        qkv_raw[(size_t)bidx * 65536 + (size_t)o * 64 + j] * sc + sh);
    srow[c][j] = qk_raw[(size_t)bg * 4096 + e] * gsc + gsh;  // row i=c
  }
  __syncthreads();

  if (tid < 64) {  // one row of the 64x64 softmax per lane
    const int i = tid;
    float mx = -3.4e38f;
    for (int j = 0; j < 64; ++j) mx = fmaxf(mx, srow[i][j]);
    float s = 0.f;
    for (int j = 0; j < 64; ++j) {
      const float ex = __expf(srow[i][j] - mx);
      srow[i][j] = ex;
      s += ex;
    }
    const float inv = 1.f / s;
    for (int j = 0; j < 64; ++j) simh[i][j] = (_Float16)(srow[i][j] * inv);
  }
  __syncthreads();

  const int wave = tid >> 5, lane = tid & 31;
  const int half = lane >> 4, ln = lane & 15, mh = half * 8;
  float* outbg = sv_raw + (size_t)bidx * 32768 + (size_t)g * 4096;
  for (int t = wave; t < 16; t += 8) {
    const int c0 = (t >> 2) * 16, i0 = (t & 3) * 16;
    v8f acc = {};
#pragma unroll
    for (int kc = 0; kc < 2; ++kc) {
      const int koff = kc * 32;
      const v16h a  = load_a_f16(&vrows[c0 + ln][koff], half);
      const v16h bm = load_b_rowk(&simh[i0 + ln][koff + half * 16]);
      acc = wmma16(a, bm, acc);
    }
#pragma unroll
    for (int r = 0; r < 8; ++r)
      outbg[(size_t)(c0 + mh + r) * 64 + i0 + ln] = acc[r];
  }
}

// ---------------------------------------------------------------------------
// Kernel 5: out[n][c][h][w] = BN_c(sv[b=n*64+w][c][h]) with LDS transpose so
// both the global read (along h) and write (along w) are contiguous.
__global__ __launch_bounds__(256) void k_out(
    const float* __restrict__ sv_raw, const float* __restrict__ st_sv,
    const float* __restrict__ g_out, const float* __restrict__ b_out,
    float* __restrict__ out) {
  __shared__ float tile[64][65];
  const int nc = blockIdx.x;
  const int nimg = nc >> 9, c = nc & 511;
  const int tid = threadIdx.x;
  float sc, sh;
  bn_coef(st_sv, g_out, b_out, c, 65536.f, &sc, &sh);
  for (int e = tid; e < 4096; e += 256) {
    const int w = e >> 6, h = e & 63;
    tile[w][h] =
        sv_raw[(size_t)(nimg * 64 + w) * 32768 + (size_t)c * 64 + h] * sc + sh;
  }
  __syncthreads();
  float* ob = out + ((size_t)nimg * 512 + c) * 4096;
  for (int e = tid; e < 4096; e += 256) {
    const int h = e >> 6, w = e & 63;
    ob[(size_t)h * 64 + w] = tile[w][h];
  }
}

// ---------------------------------------------------------------------------
extern "C" void kernel_launch(void* const* d_in, const int* in_sizes, int n_in,
                              void* d_out, int out_size, void* d_ws,
                              size_t ws_size, hipStream_t stream) {
  const float* x     = (const float*)d_in[0];
  const float* w_qkv = (const float*)d_in[1];
  const float* g_qkv = (const float*)d_in[2];
  const float* b_qkv = (const float*)d_in[3];
  const float* g_sim = (const float*)d_in[4];
  const float* b_sim = (const float*)d_in[5];
  const float* g_out = (const float*)d_in[6];
  const float* b_out = (const float*)d_in[7];
  float* out = (float*)d_out;

  // Workspace layout (floats):
  float* qkv_raw = (float*)d_ws;                  // 1024*1024*64   = 256 MB
  float* qk_raw  = qkv_raw + 67108864ull;         // 1024*8*64*64   = 128 MB
  float* sv_raw  = qk_raw + 33554432ull;          // 1024*512*64    = 128 MB
  float* st_qkv  = sv_raw + 33554432ull;          // 1024 ch * 2
  float* st_qk   = st_qkv + 2048;                 // 8 ch * 2
  float* st_sv   = st_qk + 16;                    // 512 ch * 2
  _Float16* wh   = (_Float16*)(st_sv + 1024);     // 1024*512 f16   = 1 MB

  k_zero<<<dim3(13), dim3(256), 0, stream>>>(st_qkv, 2048 + 16 + 1024);
  k_cvt_w<<<dim3(2048), dim3(256), 0, stream>>>(w_qkv, wh, 1024 * 512);

  // qkv projection (WMMA GEMM, f16 in / f32 acc)
  k_qkv_gemm<<<dim3(1024), dim3(256), 0, stream>>>(x, wh, qkv_raw);

  // BN stats for qkv: 1024 channels, per-ch (b=1024) x (h=64)
  k_bn_stats<<<dim3(1024, 8), dim3(256), 0, stream>>>(qkv_raw, st_qkv, 6, 64,
                                                      65536, 65536);

  // qk = q^T k per (b, g), fused qkv-BN
  k_qk<<<dim3(8192), dim3(256), 0, stream>>>(qkv_raw, st_qkv, g_qkv, b_qkv,
                                             qk_raw);

  // BN stats for qk: 8 groups, per-group (b=1024) x (64*64)
  k_bn_stats<<<dim3(8, 64), dim3(256), 0, stream>>>(qk_raw, st_qk, 12, 4096,
                                                    32768, 1024 * 4096);

  // softmax(BN(qk)) and sv = sim * v^T per (b, g)
  k_softmax_sv<<<dim3(8192), dim3(256), 0, stream>>>(
      qkv_raw, qk_raw, st_qkv, g_qkv, b_qkv, st_qk, g_sim, b_sim, sv_raw);

  // BN stats for sv: 512 channels, per-ch (b=1024) x (h=64)
  k_bn_stats<<<dim3(512, 8), dim3(256), 0, stream>>>(sv_raw, st_sv, 6, 64,
                                                     32768, 65536);

  // final BN + transpose to (N, C, H, W)
  k_out<<<dim3(8192), dim3(256), 0, stream>>>(sv_raw, st_sv, g_out, b_out,
                                              out);
}
